// CNN_2D_16045997817862
// MI455X (gfx1250) — matrix operand order repaired and checked by reading, not compiled, not run on hardware
//
#include <hip/hip_runtime.h>
#include <hip/hip_bf16.h>

// ---------------- types for CDNA5 WMMA ----------------
typedef __bf16 bf16_t;
typedef bf16_t v16bf __attribute__((ext_vector_type(16)));
typedef bf16_t v8bf  __attribute__((ext_vector_type(8)));
typedef unsigned short v16us __attribute__((ext_vector_type(16)));
typedef unsigned short v8us  __attribute__((ext_vector_type(8)));
typedef float v8f __attribute__((ext_vector_type(8)));

#define G 250
#define GG (G * G)           // 62500
#define NB 8                 // batch
#define C1C 16
#define C2C 32
#define HL 64
#define CONV_OUT (C2C * GG)  // 2,000,000
#define FC1_IN (CONV_OUT + 64)
#define TWO_PI 6.283185307179586f

__device__ __forceinline__ unsigned short f2bf(float f) {
    unsigned int u = __builtin_bit_cast(unsigned int, f);
    u += 0x7FFFu + ((u >> 16) & 1u);   // round-to-nearest-even
    return (unsigned short)(u >> 16);
}

// ---------------- 0) zero scratch regions ----------------
__global__ void zero_kernel(float* p, int n) {
    int i = blockIdx.x * blockDim.x + threadIdx.x;
    if (i < n) p[i] = 0.0f;
}

// ---------------- 1) rasterize lidar -> [B,250,250] f32 grid ----------------
__global__ void raster_kernel(const float* __restrict__ lidar, float* __restrict__ grid) {
    int t = blockIdx.x * blockDim.x + threadIdx.x;
    if (t >= NB * 360) return;
    int b = t / 360, a = t % 360;
    float r = lidar[b * 360 + a];
    if (!(r > 0.0f && r < 1.0f)) return;
    float ang = (TWO_PI / 359.0f) * (float)a;
    float ca = __cosf(ang) * 100.0f;
    float sa = __sinf(ang) * 100.0f;
    int n = (int)floorf((2.0f - r) * 500.0f) + 1;
    float step = (2.0f - r) / (float)(n - 1 > 1 ? n - 1 : 1);
    float* g = grid + (size_t)b * GG;
    for (int k = 0; k < n; k++) {
        float mag = r + step * (float)k;
        int xg = (int)(125.0f + mag * ca);
        int yg = (int)(125.0f + mag * sa);
        if (xg >= 0 && xg < G && yg >= 0 && yg < G) g[yg * G + xg] = 1.0f;
    }
}

// ---------------- 2) wind MLP: [B,2] -> [B,64] ----------------
__global__ void wind_kernel(const float* __restrict__ wind,
                            const float* __restrict__ w1, const float* __restrict__ b1,
                            const float* __restrict__ w2, const float* __restrict__ b2,
                            const float* __restrict__ w3, const float* __restrict__ b3,
                            float* __restrict__ wnd) {
    __shared__ float h[64];
    int o = threadIdx.x;   // 64 threads
    for (int b = 0; b < NB; b++) {
        float wx = wind[b * 2 + 0], wy = wind[b * 2 + 1];
        float v = fmaxf(w1[o * 2] * wx + w1[o * 2 + 1] * wy + b1[o], 0.0f);
        h[o] = v; __syncthreads();
        float s = b2[o];
        for (int i = 0; i < 64; i++) s += w2[o * 64 + i] * h[i];
        v = fmaxf(s, 0.0f);
        __syncthreads(); h[o] = v; __syncthreads();
        s = b3[o];
        for (int i = 0; i < 64; i++) s += w3[o * 64 + i] * h[i];
        wnd[b * 64 + o] = fmaxf(s, 0.0f);
        __syncthreads();
    }
}

// ---------------- 3) conv1 3x3 (1->16) + relu, store bf16 NHWC [b][y][x][ic] ----------------
__global__ void conv1_kernel(const float* __restrict__ grid,
                             const float* __restrict__ w, const float* __restrict__ bias,
                             unsigned short* __restrict__ h1) {
    int t = blockIdx.x * blockDim.x + threadIdx.x;
    if (t >= NB * GG) return;
    int b = t / GG, p = t % GG;
    int y = p / G, x = p % G;
    float in[9];
    const float* gb = grid + (size_t)b * GG;
#pragma unroll
    for (int ky = 0; ky < 3; ky++)
#pragma unroll
        for (int kx = 0; kx < 3; kx++) {
            int yy = y + ky - 1, xx = x + kx - 1;
            in[ky * 3 + kx] = (yy >= 0 && yy < G && xx >= 0 && xx < G) ? gb[yy * G + xx] : 0.0f;
        }
    v8us r0, r1;
#pragma unroll
    for (int oc = 0; oc < C1C; oc++) {
        float s = bias[oc];
#pragma unroll
        for (int j = 0; j < 9; j++) s += w[oc * 9 + j] * in[j];
        unsigned short hv = f2bf(fmaxf(s, 0.0f));
        if (oc < 8) r0[oc] = hv; else r1[oc - 8] = hv;
    }
    unsigned short* dst = h1 + ((size_t)b * GG + p) * C1C;   // 32B-aligned contiguous
    *(v8us*)dst = r0;
    *(v8us*)(dst + 8) = r1;
}

// ---------------- 4) conv2 3x3 (16->32) + relu via bf16 WMMA ----------------
// Block = 8 waves = 8-row x 16-col output tile. Input halo tile staged in LDS
// [10 rows][18 cols][16 ic] bf16 via GLOBAL_LOAD_ASYNC_TO_LDS_B128 (ASYNCcnt),
// overlapped with the B-fragment build; OOB halo positions zero-filled with ds
// stores to disjoint addresses (single barrier suffices).
__global__ void __launch_bounds__(256) conv2_kernel(const unsigned short* __restrict__ h1,
                                                    const float* __restrict__ w2,
                                                    const float* __restrict__ bias2,
                                                    unsigned short* __restrict__ h2) {
    __shared__ __align__(16) unsigned short sTile[10 * 18 * 16];   // 5760 B
    __shared__ __align__(16) unsigned short sB[2 * 5 * 32 * 16];   // 10240 B
    int tid = threadIdx.x;

    // ---- decode tile
    int blk = blockIdx.x;                 // NB * 32 * 16 = 4096
    int b  = blk >> 9;
    int rem = blk & 511;
    int ty = rem >> 4, tx = rem & 15;
    int y0 = ty * 8, x0 = tx * 16;

    // ---- stage input halo tile: async DMA global->LDS (issued first, waited later)
    const unsigned short* hb = h1 + (size_t)b * GG * C1C;
    for (int idx = tid; idx < 360; idx += 256) {   // 180 positions x 2 halves of 16B
        int pos = idx >> 1, half = idx & 1;
        int yy = pos / 18, xx = pos - yy * 18;
        int gy = y0 + yy - 1, gx = x0 + xx - 1;
        unsigned short* ldst = &sTile[(pos * 16) + half * 8];
        if (gy >= 0 && gy < G && gx >= 0 && gx < G) {
            const unsigned short* gsrc = hb + ((size_t)gy * G + gx) * C1C + half * 8;
            unsigned lds_off = (unsigned)(size_t)ldst;   // flat addr[31:0] == LDS offset
            asm volatile("global_load_async_to_lds_b128 %0, %1, off"
                         :: "v"(lds_off), "v"(gsrc) : "memory");
        } else {
            *(uint4*)ldst = make_uint4(0u, 0u, 0u, 0u);  // zero halo (disjoint addrs)
        }
    }

    // ---- B fragments (overlaps with async copies): B[k][n] = w2[oc][ic][ky][kx]
    for (int idx = tid; idx < 5120; idx += 256) {
        int e = idx & 15;
        int lane = (idx >> 4) & 31;
        int kc = (idx >> 9) % 5;
        int nblk = idx / 2560;
        int n = lane & 15, khalf = lane >> 4;
        int gk = kc * 32 + (e >> 3) * 16 + khalf * 8 + (e & 7);
        unsigned short v = 0;
        if (gk < 144) {
            int tap = gk >> 4, ic = gk & 15;
            v = f2bf(w2[(nblk * 16 + n) * 144 + ic * 9 + tap]);
        }
        sB[idx] = v;
    }

    asm volatile("s_wait_asynccnt 0x0" ::: "memory");
    __syncthreads();

    int lane = tid & 31, wave = tid >> 5;
    int m = lane & 15, khalf = lane >> 4;
    int y = y0 + wave;                     // this wave's output row
    // per-lane A base: tile row = wave + ky, tile col = m + kx, ic base = khalf*8
    const unsigned short* ta = &sTile[((wave * 18) + m) * 16 + khalf * 8];

    v8f c0 = {0.f,0.f,0.f,0.f,0.f,0.f,0.f,0.f};
    v8f c1 = c0;
#pragma unroll
    for (int kc = 0; kc < 5; kc++) {
        const int t0 = 2 * kc, t1 = 2 * kc + 1;
        const int ky0 = t0 / 3, kx0 = t0 % 3;
        v8us lo = *(const v8us*)(ta + (ky0 * 18 + kx0) * 16);
        v8us hi;
        if (t1 < 9) {
            const int ky1 = t1 / 3, kx1 = t1 % 3;
            hi = *(const v8us*)(ta + (ky1 * 18 + kx1) * 16);
        } else {
            hi = (v8us){0,0,0,0,0,0,0,0};
        }
        v16us au = __builtin_shufflevector(lo, hi, 0,1,2,3,4,5,6,7,8,9,10,11,12,13,14,15);
        v16bf a = __builtin_bit_cast(v16bf, au);
        v16us b0u = *(const v16us*)&sB[(0 * 5 + kc) * 512 + lane * 16];
        v16us b1u = *(const v16us*)&sB[(1 * 5 + kc) * 512 + lane * 16];
        c0 = __builtin_amdgcn_wmma_f32_16x16x32_bf16(false, a, false,
                 __builtin_bit_cast(v16bf, b0u), (short)0, c0, false, false);
        c1 = __builtin_amdgcn_wmma_f32_16x16x32_bf16(false, a, false,
                 __builtin_bit_cast(v16bf, b1u), (short)0, c1, false, false);
    }

    // ---- store: D[M][N], M = j + 8*khalf (pixel x0+M), N = lane&15 (oc); h2 is NCHW bf16
    if (y < G) {
        unsigned short* ob = h2 + (size_t)b * C2C * GG;
        int n = lane & 15;
        float bn0 = bias2[n], bn1 = bias2[16 + n];
#pragma unroll
        for (int j = 0; j < 8; j++) {
            int xo = x0 + j + khalf * 8;
            if (xo < G) {
                ob[(size_t)n * GG + y * G + xo]        = f2bf(fmaxf(c0[j] + bn0, 0.0f));
                ob[(size_t)(16 + n) * GG + y * G + xo] = f2bf(fmaxf(c1[j] + bn1, 0.0f));
            }
        }
    }
}

// ---------------- 5) fc1 GEMM via WMMA: M=8(pad16) batches, N=64, K=2,000,000 ----------------
__global__ void __launch_bounds__(256) fc1_kernel(const unsigned short* __restrict__ h2,
                                                  const float* __restrict__ W,
                                                  float* __restrict__ facc) {
    __shared__ float sacc[NB * HL];   // 512
    int tid = threadIdx.x;
    for (int i = tid; i < NB * HL; i += 256) sacc[i] = 0.0f;
    __syncthreads();

    int lane = tid & 31, wave = tid >> 5;
    int m = lane & 15, khalf = lane >> 4, n = lane & 15;
    int koff = khalf * 8;
    const int NCHUNK = CONV_OUT / 32;      // 62500
    int gw = blockIdx.x * 8 + wave;
    int NW = gridDim.x * 8;

    v8f c[4];
#pragma unroll
    for (int nb = 0; nb < 4; nb++) c[nb] = (v8f){0.f,0.f,0.f,0.f,0.f,0.f,0.f,0.f};

    for (int kc = gw; kc < NCHUNK; kc += NW) {
        int kbase = kc * 32;
        // A: activations already bf16; elements 0..7 = K koff..koff+7, 8..15 = K koff+16..+23
        v16us au;
        if (m < NB) {
            const unsigned short* pa = h2 + (size_t)m * CONV_OUT + kbase + koff;
            v8us lo = *(const v8us*)pa;
            v8us hi = *(const v8us*)(pa + 16);
            au = __builtin_shufflevector(lo, hi, 0,1,2,3,4,5,6,7,8,9,10,11,12,13,14,15);
        } else {
            au = (v16us){0,0,0,0,0,0,0,0,0,0,0,0,0,0,0,0};
        }
        v16bf a = __builtin_bit_cast(v16bf, au);
        // B: f32 weights K koff..koff+7 and koff+16..+23 per lane per n-block, packed cvt to bf16
#pragma unroll
        for (int nb = 0; nb < 4; nb++) {
            const float* pb = W + (size_t)(nb * 16 + n) * FC1_IN + kbase + khalf * 8;
            v8f f0 = *(const v8f*)pb;
            v8f f1 = *(const v8f*)(pb + 16);
            v8bf b0 = __builtin_convertvector(f0, v8bf);
            v8bf b1 = __builtin_convertvector(f1, v8bf);
            v16bf bb = __builtin_shufflevector(b0, b1, 0,1,2,3,4,5,6,7,8,9,10,11,12,13,14,15);
            c[nb] = __builtin_amdgcn_wmma_f32_16x16x32_bf16(false, a, false, bb,
                        (short)0, c[nb], false, false);
        }
    }
    // D rows 0..7 = batches (lanes 0-15, M=j); lanes 16-31 carry zero-pad rows
    if (lane < 16) {
#pragma unroll
        for (int nb = 0; nb < 4; nb++)
#pragma unroll
            for (int j = 0; j < 8; j++)
                atomicAdd(&sacc[j * HL + nb * 16 + n], c[nb][j]);
    }
    __syncthreads();
    for (int i = tid; i < NB * HL; i += 256) atomicAdd(&facc[i], sacc[i]);
}

// ---------------- 6) head: wind tail of fc1 + relu, fc2 + relu, fc3 ----------------
__global__ void head_kernel(const float* __restrict__ facc, const float* __restrict__ wnd,
                            const float* __restrict__ fc1w, const float* __restrict__ fc1b,
                            const float* __restrict__ fc2w, const float* __restrict__ fc2b,
                            const float* __restrict__ fc3w, const float* __restrict__ fc3b,
                            float* __restrict__ out) {
    __shared__ float x1[NB * HL];
    __shared__ float x2[NB * 32];
    int tid = threadIdx.x;   // 256
    for (int i = tid; i < NB * HL; i += 256) {
        int b = i >> 6, o = i & 63;
        float s = facc[i] + fc1b[o];
        const float* wrow = fc1w + (size_t)o * FC1_IN + CONV_OUT;
        for (int j = 0; j < 64; j++) s += wnd[b * 64 + j] * wrow[j];
        x1[i] = fmaxf(s, 0.0f);
    }
    __syncthreads();
    for (int i = tid; i < NB * 32; i += 256) {
        int b = i >> 5, p = i & 31;
        float s = fc2b[p];
        for (int o = 0; o < 64; o++) s += fc2w[p * 64 + o] * x1[b * 64 + o];
        x2[i] = fmaxf(s, 0.0f);
    }
    __syncthreads();
    for (int i = tid; i < NB * 200; i += 256) {
        int b = i / 200, q = i % 200;
        float s = fc3b[q];
        for (int p = 0; p < 32; p++) s += fc3w[q * 32 + p] * x2[b * 32 + p];
        out[i] = s;
    }
}

// ---------------- launch ----------------
extern "C" void kernel_launch(void* const* d_in, const int* in_sizes, int n_in,
                              void* d_out, int out_size, void* d_ws, size_t ws_size,
                              hipStream_t stream) {
    const float* lidar  = (const float*)d_in[0];
    const float* wind   = (const float*)d_in[1];
    const float* w_w1   = (const float*)d_in[2];
    const float* b_w1   = (const float*)d_in[3];
    const float* w_w2   = (const float*)d_in[4];
    const float* b_w2   = (const float*)d_in[5];
    const float* w_w3   = (const float*)d_in[6];
    const float* b_w3   = (const float*)d_in[7];
    const float* c1w    = (const float*)d_in[8];
    const float* c1b    = (const float*)d_in[9];
    const float* c2w    = (const float*)d_in[10];
    const float* c2b    = (const float*)d_in[11];
    const float* fc1w   = (const float*)d_in[12];
    const float* fc1b   = (const float*)d_in[13];
    const float* fc2w   = (const float*)d_in[14];
    const float* fc2b   = (const float*)d_in[15];
    const float* fc3w   = (const float*)d_in[16];
    const float* fc3b   = (const float*)d_in[17];
    float* out = (float*)d_out;

    char* ws = (char*)d_ws;
    float*          grid = (float*)ws;                               //  2,000,000 B
    unsigned short* h1   = (unsigned short*)(ws + 2000000);          // 16,000,000 B (NHWC)
    unsigned short* h2   = (unsigned short*)(ws + 18000000);         // 32,000,000 B (NCHW)
    float*          wnd  = (float*)(ws + 50000000);                  //      2,048 B
    float*          facc = (float*)(ws + 50002048);                  //      2,048 B

    zero_kernel<<<(NB * GG + 255) / 256, 256, 0, stream>>>(grid, NB * GG);
    zero_kernel<<<2, 256, 0, stream>>>(facc, NB * HL);
    raster_kernel<<<(NB * 360 + 255) / 256, 256, 0, stream>>>(lidar, grid);
    wind_kernel<<<1, 64, 0, stream>>>(wind, w_w1, b_w1, w_w2, b_w2, w_w3, b_w3, wnd);
    conv1_kernel<<<(NB * GG + 255) / 256, 256, 0, stream>>>(grid, c1w, c1b, h1);
    conv2_kernel<<<4096, 256, 0, stream>>>(h1, c2w, c2b, h2);
    fc1_kernel<<<256, 256, 0, stream>>>(h2, fc1w, facc);
    head_kernel<<<1, 256, 0, stream>>>(facc, wnd, fc1w, fc1b, fc2w, fc2b, fc3w, fc3b, out);
}
